// gated_tpp_57913339020009
// MI455X (gfx1250) — matrix operand order repaired and checked by reading, not compiled, not run on hardware
//
#include <hip/hip_runtime.h>
#include <hip/hip_bf16.h>
#include <math.h>

// ---------------- problem constants ----------------
#define B_   4
#define S_   1024
#define D_   256
#define TWOD 512
#define NT_  20
#define NS_  50
#define MROWS (B_*S_*NS_)   // 204800 rows in the big GEMM

typedef __attribute__((ext_vector_type(16))) _Float16 v16h;
typedef __attribute__((ext_vector_type(8)))  _Float16 v8h;
typedef __attribute__((ext_vector_type(4)))  _Float16 v4h;
typedef __attribute__((ext_vector_type(8)))  float    v8f;

// ---------------- workspace layout (bytes) ----------------
#define OFF_HVT   (0ull)                        // f16 [B][2D][S]   4 MiB  (transposed hidden_vec)
#define OFF_AS    (OFF_HVT   + 4ull*1024*1024)  // f32 [B*S]
#define OFF_BS    (OFF_AS    + 16384ull)
#define OFF_AG    (OFF_BS    + 16384ull)
#define OFF_BG    (OFF_AG    + 16384ull)
#define OFF_HID   (OFF_BG    + 16384ull)        // f32 [B*S][2D]    8 MiB
#define OFF_HLN   (OFF_HID   + 8ull*1024*1024)  // f16 [B*S][2D]    4 MiB
#define OFF_PROJ  (OFF_HLN   + 4ull*1024*1024)  // f32 [B*S][2D]    8 MiB
#define OFF_WIN16 (OFF_PROJ  + 8ull*1024*1024)  // f16 [2D][2D]     512 KiB
#define OFF_WN16  (OFF_WIN16 + 512ull*1024)     // f16 [2D][2D]     512 KiB

// ---------------- device helpers ----------------
// fast softplus: v_exp_f32 + v_log_f32 instead of libm log1pf's 40-op chain
__device__ __forceinline__ float softplus_f(float x){
  float sp = __logf(1.f + __expf(x));
  return x > 20.f ? x : sp;     // select, no branch
}

__device__ __forceinline__ v8f wmma16(v16h a, v16h b, v8f c){
  // D = A(16x32 f16) x B(32x16 f16) + C(16x16 f32)
  return __builtin_amdgcn_wmma_f32_16x16x32_f16(false, a, false, b, (short)0, c, false, false);
}

// A-fragment (16x32 f16, M x K) from an LDS tile stored row-major [16][32].
// lanes 0-15: M=lane, K 0-15;
// lanes 16-31: M=lane-16, K 16-31 (matches load_bt_glob).
__device__ __forceinline__ v16h load_a_lds(const _Float16* tile, int lane){
  int row = lane & 15;
  int kb  = (lane >> 4) << 4;          // 0 or 16
  const _Float16* p = tile + row*32 + kb;
  return *(const v16h*)(p);            // K kb..kb+15
}

// A-fragment from a global row-major f16 matrix [M][ldk]
__device__ __forceinline__ v16h load_a_glob(const _Float16* A, int m0, int k0, int ldk, int lane){
  const _Float16* p = A + (size_t)(m0 + (lane & 15)) * ldk + k0 + ((lane >> 4) << 4);
  return *(const v16h*)(p);
}

// B-fragment (32x16 f16, K x N) from "B-transposed" global BT[N][ldk] (BT[n][k]=B[k][n]).
// lane<16: N=n0+lane, K k0..k0+15; lane>=16: N=n0+lane-16, K k0+16..k0+31 -> one 32B load.
__device__ __forceinline__ v16h load_bt_glob(const _Float16* BT, int n0, int k0, int ldk, int lane){
  const _Float16* p = BT + (size_t)(n0 + (lane & 15)) * ldk + k0 + ((lane >> 4) << 4);
  return *(const v16h*)(p);
}

// C/D 16x16 f32: VGPR r holds M=r (lanes 0-15, N=lane) / M=8+r (lanes 16-31, N=lane-16)
__device__ __forceinline__ void store_c(float* C, v8f c, int m0, int n0, int ldn, int lane){
  int n  = n0 + (lane & 15);
  int mb = m0 + ((lane >> 4) << 3);
  #pragma unroll
  for (int r = 0; r < 8; ++r) C[(size_t)(mb + r) * ldn + n] = c[r];
}

__device__ __forceinline__ float block_sum_256(float v, float* buf){
  int t = threadIdx.x;
  buf[t] = v; __syncthreads();
  #pragma unroll
  for (int off = 128; off > 0; off >>= 1){
    if (t < off) buf[t] += buf[t + off];
    __syncthreads();
  }
  float r = buf[0];
  __syncthreads();
  return r;
}

// ---------------- kernel 0: zero pred_time ----------------
__global__ void k_zero(float* p, int n){
  int i = blockIdx.x * blockDim.x + threadIdx.x;
  if (i < n) p[i] = 0.f;
}

// ---------------- kernel 1: pe + temb -> hvT (f16, transposed) + rank-1 projections ----
__global__ __launch_bounds__(256) void k_prep(const int* __restrict__ etype,
    const float* __restrict__ etime, const float* __restrict__ wt,
    const float* __restrict__ table, const float* __restrict__ w_sigma,
    const float* __restrict__ w_gate,
    _Float16* __restrict__ hvT, float* a_s, float* b_s, float* a_g, float* b_g)
{
  __shared__ float rbuf[256];
  int e = blockIdx.x;
  int b = e >> 10, s = e & 1023;
  int t = threadIdx.x;
  float tm = etime[e];
  int ty = etype[e];
  const float c0 = -2.0f * 9.2103403719761836f / 256.0f; // -2*ln(1e4)/d
  float v0;
  if (t < 128){
    float dv = __expf((float)t * c0);
    v0 = __sinf((float)s * dv + tm * wt[t]);
  } else {
    int k = t - 128;
    float dv = __expf((float)k * c0);
    v0 = __cosf((float)s * dv + tm * wt[k]);
  }
  float v1 = table[(size_t)ty * D_ + t];       // type embedding
  hvT[((size_t)b * TWOD + t) * S_ + s]       = (_Float16)v0;
  hvT[((size_t)b * TWOD + t + 256) * S_ + s] = (_Float16)v1;
  float r;
  r = block_sum_256(v1 * w_sigma[t],       rbuf); if (t==0) a_s[e] = r;
  r = block_sum_256(v1 * w_sigma[256 + t], rbuf); if (t==0) b_s[e] = r;
  r = block_sum_256(v1 * w_gate[t],        rbuf); if (t==0) a_g[e] = r;
  r = block_sum_256(v1 * w_gate[256 + t],  rbuf); if (t==0) b_g[e] = r;
}

// ---------------- kernel 2: convert weight matrices to f16 ----------------
__global__ void k_cvt(const float* __restrict__ a, const float* __restrict__ b,
                      _Float16* a16, _Float16* b16, int n){
  for (int i = blockIdx.x * blockDim.x + threadIdx.x; i < n; i += gridDim.x * blockDim.x){
    a16[i] = (_Float16)a[i];
    b16[i] = (_Float16)b[i];
  }
}

// ---------------- kernel 3: fused causal scores x hidden_vec (flash-style WMMA) ----
// grid (S/16=64, B), 256 threads = 8 waves.  One block owns a 16-row tile x FULL N=512
// (wave w -> 4 C tiles at nb=w*64).  Score tiles are double-buffered in LDS: the
// transcendental math for step ks+1 overlaps the 4 WMMAs of step ks; ONE barrier/step.
__global__ __launch_bounds__(256) void k_scores(const float* __restrict__ etime,
    const float* __restrict__ a_s, const float* __restrict__ b_s,
    const float* __restrict__ a_g, const float* __restrict__ b_g,
    const _Float16* __restrict__ hvT, float* __restrict__ hidden)
{
  __shared__ __align__(32) _Float16 aT[2][16 * 32];
  int t = threadIdx.x, lane = t & 31, w = t >> 5;
  int b  = blockIdx.y;
  int i0 = blockIdx.x * 16;
  int nb = w * 64;
  const float* tt = etime + b * S_;
  const float* as = a_s + b * S_;
  const float* bs = b_s + b * S_;
  const float* ag = a_g + b * S_;
  const float* bg = b_g + b * S_;
  const _Float16* hb = hvT + (size_t)b * TWOD * S_;
  int r  = t >> 4;            // 0..15: score-tile row this thread fills
  int cb = (t & 15) * 2;      // 2 columns per thread
  int i  = i0 + r;
  float ti = tt[i], bsv = bs[i], bgv = bg[i];

  auto score_tile = [&](int j0, _Float16* dst){
    #pragma unroll
    for (int u = 0; u < 2; ++u){
      int j = j0 + cb + u;
      float sig = softplus_f(as[j] + bsv);
      float gt  = 1.f / (1.f + __expf(-(ag[j] + bgv)));
      float td  = fabsf(tt[j] - ti);
      float sc  = gt * __expf(-sig * td);
      sc = (j < i) ? sc : 0.f;             // causal, branchless
      dst[r * 32 + cb + u] = (_Float16)sc;
    }
  };

  v8f c0 = {}, c1 = {}, c2 = {}, c3 = {};
  int nsteps = i0 / 32 + 1;   // causal: only K-tiles with j <= i0+14 contribute
  score_tile(0, aT[0]);
  __syncthreads();
  for (int ks = 0; ks < nsteps; ++ks){
    int j0 = ks * 32;
    const _Float16* cur = aT[ks & 1];
    if (ks + 1 < nsteps) score_tile(j0 + 32, aT[(ks + 1) & 1]);
    // hoist all B loads so the scheduler can clause them and stagger waits
    v16h af = load_a_lds(cur, lane);
    v16h b0 = load_bt_glob(hb, nb,      j0, S_, lane);
    v16h b1 = load_bt_glob(hb, nb + 16, j0, S_, lane);
    v16h b2 = load_bt_glob(hb, nb + 32, j0, S_, lane);
    v16h b3 = load_bt_glob(hb, nb + 48, j0, S_, lane);
    c0 = wmma16(af, b0, c0);
    c1 = wmma16(af, b1, c1);
    c2 = wmma16(af, b2, c2);
    c3 = wmma16(af, b3, c3);
    __syncthreads();
  }
  float* Cb = hidden + (size_t)b * S_ * TWOD;
  store_c(Cb, c0, i0, nb,      TWOD, lane);
  store_c(Cb, c1, i0, nb + 16, TWOD, lane);
  store_c(Cb, c2, i0, nb + 32, TWOD, lane);
  store_c(Cb, c3, i0, nb + 48, TWOD, lane);
}

// ---------------- kernel 4: LayerNorm + mark softmax + f16 hidden_ln ----------------
__global__ __launch_bounds__(256) void k_post(const float* __restrict__ hidden,
    const float* __restrict__ gamma, const float* __restrict__ beta,
    const float* __restrict__ w_pred, _Float16* __restrict__ hln,
    float* __restrict__ out_mark)
{
  __shared__ float rbuf[256];
  __shared__ float gbuf[TWOD];
  __shared__ float lgs[NT_];
  __shared__ float ms[2];
  int e = blockIdx.x, t = threadIdx.x;
  float x0 = hidden[(size_t)e * TWOD + t];
  float x1 = hidden[(size_t)e * TWOD + t + 256];
  float sum = block_sum_256(x0 + x1, rbuf);
  float ssq = block_sum_256(x0 * x0 + x1 * x1, rbuf);
  float mu   = sum * (1.f / 512.f);
  float var  = ssq * (1.f / 512.f) - mu * mu;
  float rstd = rsqrtf(var + 1e-6f);
  float g0 = (x0 - mu) * rstd * gamma[t]       + beta[t];
  float g1 = (x1 - mu) * rstd * gamma[t + 256] + beta[t + 256];
  hln[(size_t)e * TWOD + t]       = (_Float16)g0;
  hln[(size_t)e * TWOD + t + 256] = (_Float16)g1;
  gbuf[t] = g0; gbuf[t + 256] = g1;
  __syncthreads();
  if (t < NT_){
    float acc = 0.f;
    const float* wp = w_pred + (size_t)t * TWOD;
    for (int h = 0; h < TWOD; ++h) acc += gbuf[h] * wp[h];
    lgs[t] = acc;
  }
  __syncthreads();
  if (t == 0){
    float m = lgs[0];
    for (int n = 1; n < NT_; ++n) m = fmaxf(m, lgs[n]);
    float sm = 0.f;
    for (int n = 0; n < NT_; ++n) sm += __expf(lgs[n] - m);
    ms[0] = m; ms[1] = sm;
  }
  __syncthreads();
  if (t < NT_) out_mark[(size_t)e * NT_ + t] = __expf(lgs[t] - ms[0]) / ms[1];
}

// ---------------- kernel 5: proj = hidden_ln @ w_in^T  (WMMA f16) ----------------
__global__ __launch_bounds__(256) void k_proj(const _Float16* __restrict__ hln,
    const _Float16* __restrict__ win16, float* __restrict__ proj)
{
  int t = threadIdx.x, lane = t & 31, w = t >> 5;
  int m0 = blockIdx.y * 16;
  int n0 = blockIdx.x * 128 + w * 16;
  v8f c = {};
  for (int k0 = 0; k0 < TWOD; k0 += 32){
    v16h af = load_a_glob(hln, m0, k0, TWOD, lane);
    v16h bf = load_bt_glob(win16, n0, k0, TWOD, lane);
    c = wmma16(af, bf, c);
  }
  store_c(proj, c, m0, n0, TWOD, lane);
}

// ---------------- kernel 6: fused noise GEMM + relu + .w_time + softplus + mean ----
// grid = MROWS/64 blocks, 256 threads = 8 waves.  Block: 64 rows (4 M-subtiles) x full
// N=512.  Wave w owns 4 N tiles at nb=w*64; each B fragment is loaded ONCE per K-step
// and reused across 4 M-subtiles (16 WMMAs/wave/step).  The f32->f16 A staging is
// double-buffered in LDS so conversion of step ks+1 overlaps the WMMAs of step ks;
// ONE barrier per step.  h[B,S,NS,2D] (420 MB) is never materialized.
__global__ __launch_bounds__(256) void k_big(const float* __restrict__ noise,
    const _Float16* __restrict__ wn16, const float* __restrict__ proj,
    const float* __restrict__ w_time, float* __restrict__ pred)
{
  __shared__ __align__(32) _Float16 aT[2][64 * 32];   // 2 x 4 KiB
  __shared__ float wts[TWOD];
  __shared__ float red[64];
  int t = threadIdx.x, lane = t & 31, w = t >> 5;
  int m0 = blockIdx.x * 64;
  wts[t] = w_time[t]; wts[t + 256] = w_time[t + 256];
  if (t < 64) red[t] = 0.f;
  // A staging: 64x32 f32->f16, 8 contiguous elements per thread
  int abase = t * 8;
  int arow  = abase >> 5;        // 0..63
  int acol  = abase & 31;        // multiple of 8
  const float* asrc = noise + (size_t)(m0 + arow) * TWOD + acol;

  auto stage = [&](int k0, _Float16* dstbase){
    float4 f0 = *(const float4*)(asrc + k0);
    float4 f1 = *(const float4*)(asrc + k0 + 4);
    v4h h0 = { (_Float16)f0.x, (_Float16)f0.y, (_Float16)f0.z, (_Float16)f0.w };
    v4h h1 = { (_Float16)f1.x, (_Float16)f1.y, (_Float16)f1.z, (_Float16)f1.w };
    _Float16* dst = dstbase + arow * 32 + acol;
    *(v4h*)(dst)     = h0;
    *(v4h*)(dst + 4) = h1;
  };

  int nb = w * 64;
  v8f cc[4][4];                  // [m-subtile][n-tile]
  #pragma unroll
  for (int ms = 0; ms < 4; ++ms)
    #pragma unroll
    for (int q = 0; q < 4; ++q) cc[ms][q] = (v8f){};
  stage(0, aT[0]);
  __syncthreads();
  for (int ks = 0; ks < TWOD / 32; ++ks){
    int k0 = ks * 32;
    const _Float16* cur = aT[ks & 1];
    if (ks + 1 < TWOD / 32){
      __builtin_prefetch(asrc + k0 + 64, 0, 0);     // stream A two steps ahead
      stage(k0 + 32, aT[(ks + 1) & 1]);
    }
    v16h b0 = load_bt_glob(wn16, nb,      k0, TWOD, lane);
    v16h b1 = load_bt_glob(wn16, nb + 16, k0, TWOD, lane);
    v16h b2 = load_bt_glob(wn16, nb + 32, k0, TWOD, lane);
    v16h b3 = load_bt_glob(wn16, nb + 48, k0, TWOD, lane);
    #pragma unroll
    for (int ms = 0; ms < 4; ++ms){
      v16h af = load_a_lds(cur + ms * (16 * 32), lane);
      cc[ms][0] = wmma16(af, b0, cc[ms][0]);
      cc[ms][1] = wmma16(af, b1, cc[ms][1]);
      cc[ms][2] = wmma16(af, b2, cc[ms][2]);
      cc[ms][3] = wmma16(af, b3, cc[ms][3]);
    }
    __syncthreads();
  }
  // epilogue: v = relu(h + proj[e,n]); partial[m] += v * w_time[n]
  int half = lane >> 4, ncol = lane & 15;
  #pragma unroll
  for (int ms = 0; ms < 4; ++ms){
    float part[8];
    #pragma unroll
    for (int r = 0; r < 8; ++r) part[r] = 0.f;
    #pragma unroll
    for (int q = 0; q < 4; ++q){
      int n = nb + q * 16 + ncol;
      float wv = wts[n];
      #pragma unroll
      for (int r = 0; r < 8; ++r){
        int mloc = ms * 16 + half * 8 + r;
        int e = (m0 + mloc) / NS_;
        float v = cc[ms][q][r] + proj[(size_t)e * TWOD + n];
        v = v > 0.f ? v : 0.f;
        part[r] += v * wv;
      }
    }
    #pragma unroll
    for (int r = 0; r < 8; ++r)
      atomicAdd(&red[ms * 16 + half * 8 + r], part[r]);
  }
  __syncthreads();
  if (t < 64){
    int e = (m0 + t) / NS_;
    float sp = softplus_f(red[t]);
    atomicAdd(&pred[e], sp * (1.0f / NS_));
  }
}

// ---------------- host launcher ----------------
extern "C" void kernel_launch(void* const* d_in, const int* in_sizes, int n_in,
                              void* d_out, int out_size, void* d_ws, size_t ws_size,
                              hipStream_t stream) {
  const int*   event_type = (const int*)  d_in[0];
  const float* event_time = (const float*)d_in[1];
  const float* noise      = (const float*)d_in[2];
  const float* wt         = (const float*)d_in[3];
  const float* table      = (const float*)d_in[4];
  const float* w_sigma    = (const float*)d_in[5];
  const float* w_gate     = (const float*)d_in[6];
  const float* ln_gamma   = (const float*)d_in[7];
  const float* ln_beta    = (const float*)d_in[8];
  const float* w_pred     = (const float*)d_in[9];
  const float* w_in       = (const float*)d_in[10];
  const float* w_noise    = (const float*)d_in[11];
  const float* w_time     = (const float*)d_in[12];

  float* out_pred = (float*)d_out;              // [B*S]
  float* out_mark = (float*)d_out + B_ * S_;    // [B*S*NT]

  char* ws = (char*)d_ws;
  _Float16* hvT   = (_Float16*)(ws + OFF_HVT);
  float*    a_s   = (float*)(ws + OFF_AS);
  float*    b_s   = (float*)(ws + OFF_BS);
  float*    a_g   = (float*)(ws + OFF_AG);
  float*    b_g   = (float*)(ws + OFF_BG);
  float*    hid   = (float*)(ws + OFF_HID);
  _Float16* hln   = (_Float16*)(ws + OFF_HLN);
  float*    proj  = (float*)(ws + OFF_PROJ);
  _Float16* win16 = (_Float16*)(ws + OFF_WIN16);
  _Float16* wn16  = (_Float16*)(ws + OFF_WN16);

  k_zero<<<(B_ * S_ + 255) / 256, 256, 0, stream>>>(out_pred, B_ * S_);
  k_prep<<<B_ * S_, 256, 0, stream>>>(event_type, event_time, wt, table,
                                      w_sigma, w_gate, hvT, a_s, b_s, a_g, b_g);
  k_cvt<<<512, 256, 0, stream>>>(w_in, w_noise, win16, wn16, TWOD * TWOD);
  k_scores<<<dim3(S_ / 16, B_), 256, 0, stream>>>(event_time, a_s, b_s,
                                                  a_g, b_g, hvT, hid);
  k_post<<<B_ * S_, 256, 0, stream>>>(hid, ln_gamma, ln_beta, w_pred, hln, out_mark);
  k_proj<<<dim3(TWOD / 128, (B_ * S_) / 16), 256, 0, stream>>>(hln, win16, proj);
  k_big<<<MROWS / 64, 256, 0, stream>>>(noise, wn16, proj, w_time, out_pred);
}
